// AnomalyDetector_56315611185749
// MI455X (gfx1250) — matrix operand order repaired and checked by reading, not compiled, not run on hardware
//
#include <hip/hip_runtime.h>
#include <math.h>

// ---------------------------------------------------------------------------
// AnomalyDetector on MI455X (gfx1250, wave32, WMMA + TDM).
// All GEMMs use v_wmma_f32_16x16x32_bf16 (bf16 A/B, f32 accumulate).
// B (weight) matrices are pre-swizzled into fragment-ready 32B/lane blocks so
// a B-fragment is 2x global_load_b128; A-fragments are 2x ds_load_b128.
// LSTM x_t tiles are staged into LDS by the Tensor Data Mover.
// ---------------------------------------------------------------------------

typedef __bf16 bf16;
typedef __attribute__((ext_vector_type(16))) __bf16 bf16x16;
typedef __attribute__((ext_vector_type(8)))  __bf16 bf16x8;
typedef __attribute__((ext_vector_type(4)))  __bf16 bf16x4;
typedef __attribute__((ext_vector_type(8)))  float   f32x8;
typedef __attribute__((ext_vector_type(4)))  unsigned int u32x4;
typedef __attribute__((ext_vector_type(8)))  int     i32x8;
typedef __attribute__((ext_vector_type(4)))  int     i32x4;

#define B_TOT 16384
#define T_SEQ 64
#define HID   128
#define LAT   64

static __device__ __forceinline__ f32x8 wmma_bf(bf16x16 a, bf16x16 b, f32x8 c) {
  return __builtin_amdgcn_wmma_f32_16x16x32_bf16(false, a, false, b, (short)0, c,
                                                 false, false);
}

// A fragment (16x32 bf16, row-major, leading dim lda, lda % 32 == 0).
// ISA 7.12.2: lane (m,hh) holds K = {hh*8..hh*8+7} and {16+hh*8..16+hh*8+7}
// -> two contiguous 16-byte runs -> 2x ds_load_b128.
static __device__ __forceinline__ bf16x16 load_a_frag(const bf16* A, int lda) {
  int lane = threadIdx.x & 31;
  int m = lane & 15, hh = lane >> 4;
  const bf16* p = A + m * lda + hh * 8;
  bf16x8 lo = *(const bf16x8*)(p);
  bf16x8 hi = *(const bf16x8*)(p + 16);
  return __builtin_shufflevector(lo, hi, 0, 1, 2, 3, 4, 5, 6, 7, 8, 9, 10, 11,
                                 12, 13, 14, 15);
}

// B fragment from pre-swizzled weights: block (kc,nt) holds 32 lanes x 16 bf16
// contiguously; lane L reads 32 contiguous bytes -> 2x global_load_b128.
static __device__ __forceinline__ bf16x16 load_b_packed(const bf16* P, int ntCount,
                                                        int kc, int nt) {
  int lane = threadIdx.x & 31;
  return *(const bf16x16*)(P + ((size_t)(kc * ntCount + nt) * 32 + lane) * 16);
}

static __device__ __forceinline__ float sigm(float x) { return 1.f / (1.f + __expf(-x)); }
// tanh(x) = 2*sigmoid(2x) - 1   (single v_exp_f32 + rcp instead of libm tanhf)
static __device__ __forceinline__ float tanh_fast(float x) {
  return __builtin_fmaf(2.f, 1.f / (1.f + __expf(-2.f * x)), -1.f);
}

// ---------------------------------------------------------------------------
// TDM: load a 64x16 bf16 tile (row stride 1024 elements in memory) into LDS
// with an 8-dword pad after every 8 stored dwords -> 64-byte LDS row pitch
// (i.e. the 32-column padded sX layout; pad columns are never written).
// D# layout per CDNA5 ISA ch.8 (group0: count/lds/global addr/type=2;
// group1: data_size=2B, pad_enable, pad_interval=8dw, pad_amount=8dw,
// tensor_dim0=16, tensor_dim1=64, tile 16x64, dim0_stride=1024).
// ---------------------------------------------------------------------------
#if __has_builtin(__builtin_amdgcn_tensor_load_to_lds)
#define HAVE_TDM 1
static __device__ __forceinline__ void tdm_load_x_tile(unsigned long long gaddr,
                                                       unsigned int lds_off) {
  u32x4 g0;
  g0[0] = 1u;                                                  // count=1 (user D#)
  g0[1] = lds_off;                                             // lds_addr
  g0[2] = (unsigned int)gaddr;                                 // global_addr[31:0]
  g0[3] = (unsigned int)((gaddr >> 32) & 0x01FFFFFFu) | 0x80000000u;  // ga[56:32]|type=2
  i32x8 g1;
  g1[0] = (int)((1u << 16) | (1u << 20) | (2u << 22) | (7u << 25));
  g1[1] = (int)(16u << 16);   // tensor_dim0 = 16
  g1[2] = (int)(64u << 16);   // tensor_dim1 = 64
  g1[3] = (int)(16u << 16);   // tile_dim0 = 16
  g1[4] = 64;                 // tile_dim1 = 64 (tile_dim2 = 0)
  g1[5] = 1024;               // tensor_dim0_stride = T*16 elements
  g1[6] = 0;
  g1[7] = 0;
  i32x4 z4 = {0, 0, 0, 0};
#if __clang_major__ >= 23
  i32x8 z8 = {0, 0, 0, 0, 0, 0, 0, 0};
  __builtin_amdgcn_tensor_load_to_lds(g0, g1, z4, z4, z8, 0);
#else
  __builtin_amdgcn_tensor_load_to_lds(g0, g1, z4, z4, 0);
#endif
}
#else
#define HAVE_TDM 0
#endif

// ---------------------------------------------------------------------------
// fp32 -> bf16 conversion with zero padding (row-major, for A-side activations)
// ---------------------------------------------------------------------------
__global__ void convert_pad_kernel(const float* __restrict__ src, bf16* __restrict__ dst,
                                   int rows, int cols, int dstRows, int dstCols) {
  int i = blockIdx.x * blockDim.x + threadIdx.x;
  int total = dstRows * dstCols;
  if (i >= total) return;
  int r = i / dstCols, c = i % dstCols;
  float v = (r < rows && c < cols) ? src[(size_t)r * cols + c] : 0.f;
  dst[i] = (bf16)v;
}

// ---------------------------------------------------------------------------
// fp32 [rows x cols] -> bf16 WMMA-B fragment blocks, padded to [Kpad x Npad].
// ---------------------------------------------------------------------------
__global__ void pack_b_kernel(const float* __restrict__ src, bf16* __restrict__ dst,
                              int rows, int cols, int Kpad, int Npad) {
  int i = blockIdx.x * blockDim.x + threadIdx.x;
  int ntCount = Npad >> 4;
  int total = (Kpad >> 5) * ntCount * 32;
  if (i >= total) return;
  int lane = i & 31;
  int t = i >> 5;
  int nt = t % ntCount;
  int kc = t / ntCount;
  int n = lane & 15, hh = lane >> 4;
  int col = nt * 16 + n;
  bf16* d = dst + (size_t)i * 16;
#pragma unroll
  for (int j = 0; j < 8; ++j) {
    int k = kc * 32 + (hh << 3) + ((j & 4) << 2) + ((j & 3) << 1);
    d[2 * j]     = (bf16)((k < rows && col < cols) ? src[(size_t)k * cols + col] : 0.f);
    d[2 * j + 1] = (bf16)((k + 1 < rows && col < cols) ? src[(size_t)(k + 1) * cols + col] : 0.f);
  }
}

__global__ void add_vec_kernel(const float* __restrict__ a, const float* __restrict__ b,
                               float* __restrict__ dst, int n) {
  int i = blockIdx.x * blockDim.x + threadIdx.x;
  if (i < n) dst[i] = a[i] + b[i];
}

// ---------------------------------------------------------------------------
// log|det| of a 64x64 matrix (one block per flow, 64 threads, LU w/ pivoting)
// ---------------------------------------------------------------------------
__global__ void slogdet64_kernel(const float* __restrict__ W0, const float* __restrict__ W1,
                                 const float* __restrict__ W2, float* __restrict__ out3) {
  __shared__ float M[64 * 64];
  __shared__ float s_ld;
  __shared__ int s_piv;
  const float* W = (blockIdx.x == 0) ? W0 : ((blockIdx.x == 1) ? W1 : W2);
  int tid = threadIdx.x;
  for (int i = tid; i < 4096; i += 64) M[i] = W[i];
  if (tid == 0) s_ld = 0.f;
  __syncthreads();
  for (int k = 0; k < 64; ++k) {
    if (tid == 0) {
      int piv = k; float best = fabsf(M[k * 64 + k]);
      for (int i = k + 1; i < 64; ++i) {
        float v = fabsf(M[i * 64 + k]);
        if (v > best) { best = v; piv = i; }
      }
      s_piv = piv;
      s_ld += logf(best);
    }
    __syncthreads();
    int piv = s_piv;
    if (piv != k) { float t = M[k * 64 + tid]; M[k * 64 + tid] = M[piv * 64 + tid]; M[piv * 64 + tid] = t; }
    __syncthreads();
    if (tid > k) {
      float f = M[tid * 64 + k] / M[k * 64 + k];
      for (int c = k; c < 64; ++c) M[tid * 64 + c] -= f * M[k * 64 + c];
    }
    __syncthreads();
  }
  if (tid == 0) out3[blockIdx.x] = s_ld;
}

// ---------------------------------------------------------------------------
// Generic 2-layer MLP: out = act2( relu(X@W1+b1) @ W2 + b2 )
// ---------------------------------------------------------------------------
template <int KC>
__global__ __launch_bounds__(128) void mlp_kernel(
    const bf16* __restrict__ X, const bf16* __restrict__ W1p,
    const float* __restrict__ b1, const bf16* __restrict__ W2p,
    const float* __restrict__ b2, float* __restrict__ out, int O, int Opad,
    bf16* __restrict__ out_bf, int act) {
  constexpr int LDX = KC * 32;
  __shared__ __align__(16) bf16 sX[64 * LDX];
  __shared__ __align__(16) bf16 sH[64 * HID];
  int tid = threadIdx.x, wave = tid >> 5, lane = tid & 31;
  int rowBase = blockIdx.x * 64;
  int n = lane & 15, hh = lane >> 4;

  const size_t base = (size_t)rowBase * LDX;
  for (int i = tid; i < 64 * LDX; i += 128) sX[i] = X[base + i];
  __syncthreads();

  const bf16* Aw = sX + (wave * 16) * LDX;
  bf16x16 aF[KC];
#pragma unroll
  for (int kc = 0; kc < KC; ++kc) aF[kc] = load_a_frag(Aw + kc * 32, LDX);

#pragma unroll
  for (int nt = 0; nt < 8; ++nt) {
    f32x8 acc = {0.f, 0.f, 0.f, 0.f, 0.f, 0.f, 0.f, 0.f};
#pragma unroll
    for (int kc = 0; kc < KC; ++kc) {
      bf16x16 b = load_b_packed(W1p, 8, kc, nt);
      acc = wmma_bf(aF[kc], b, acc);
    }
    int col = nt * 16 + n;
    float bb = b1[col];
#pragma unroll
    for (int j = 0; j < 8; ++j) {
      float v = acc[j] + bb;
      v = v > 0.f ? v : 0.f;
      sH[(wave * 16 + j + 8 * hh) * HID + col] = (bf16)v;
    }
  }
  const bf16* Ah = sH + (wave * 16) * HID;
  bf16x16 hF[4];
#pragma unroll
  for (int kc = 0; kc < 4; ++kc) hF[kc] = load_a_frag(Ah + kc * 32, HID);
  int ntCount = Opad >> 4;
  for (int nt = 0; nt < ntCount; ++nt) {
    f32x8 acc = {0.f, 0.f, 0.f, 0.f, 0.f, 0.f, 0.f, 0.f};
#pragma unroll
    for (int kc = 0; kc < 4; ++kc) {
      bf16x16 b = load_b_packed(W2p, ntCount, kc, nt);
      acc = wmma_bf(hF[kc], b, acc);
    }
    int col = nt * 16 + n;
    if (col < O) {
      float bb = b2[col];
#pragma unroll
      for (int j = 0; j < 8; ++j) {
        int rowg = rowBase + wave * 16 + j + 8 * hh;
        float v = acc[j] + bb;
        if (act == 1) v = sigm(v);
        out[(size_t)rowg * O + col] = v;
        if (out_bf) out_bf[(size_t)rowg * LAT + col] = (bf16)v;
      }
    }
  }
}

// ---------------------------------------------------------------------------
// Fused 2-layer LSTM (torch gate order i,f,g,o) + temporal_proj epilogue.
// x_t staged into LDS by the Tensor Data Mover (wave 0 issues, all wait).
// A zero offset laundered through empty asm each step keeps the weight
// B-fragment loads loop-variant (no LICM spill) while preserving the global
// address space of the kernarg pointers.
// ---------------------------------------------------------------------------
__global__ __launch_bounds__(128) void lstm_fused_kernel(
    const bf16* __restrict__ xtb,                       // [B,T,16] bf16
    const bf16* __restrict__ Wih0p,                     // packed [32 x 512]
    const bf16* __restrict__ Whh0p,                     // packed [128 x 512]
    const float* __restrict__ bias0,                    // [512] = bih+bhh
    const bf16* __restrict__ Wih1p,                     // packed [128 x 512]
    const bf16* __restrict__ Whh1p,                     // packed [128 x 512]
    const float* __restrict__ bias1,                    // [512]
    const bf16* __restrict__ Wprojp,                    // packed [128 x 64]
    const float* __restrict__ bproj,                    // [64]
    float* __restrict__ enc_out,                        // [B x 64]
    bf16* __restrict__ enc_bf) {                        // [B x 64]
  __shared__ __align__(16) bf16 sX[64 * 32];
  __shared__ __align__(16) bf16 sH1[64 * HID];
  __shared__ __align__(16) bf16 sH2[64 * HID];
  __shared__ float sB[1024];                            // bias0 | bias1
  int tid = threadIdx.x, wave = tid >> 5, lane = tid & 31;
  int rowBase = blockIdx.x * 64;
  int n = lane & 15, hh = lane >> 4;

  float c1[8][8], c2[8][8];
#pragma unroll
  for (int a = 0; a < 8; ++a)
#pragma unroll
    for (int b = 0; b < 8; ++b) { c1[a][b] = 0.f; c2[a][b] = 0.f; }
  for (int i = tid; i < 64 * HID; i += 128) { sH1[i] = (bf16)0.f; sH2[i] = (bf16)0.f; }
  for (int i = tid; i < 64 * 32; i += 128) sX[i] = (bf16)0.f;  // pad cols stay 0
  for (int i = tid; i < 512; i += 128) { sB[i] = bias0[i]; sB[512 + i] = bias1[i]; }

  const bf16* Ax  = sX  + (wave * 16) * 32;
  const bf16* A1  = sH1 + (wave * 16) * HID;
  const bf16* A2s = sH2 + (wave * 16) * HID;
  int sh1Row = (wave * 16 + 8 * hh) * HID;  // base row offset for h stores
  int sh2Row = sh1Row;

#if HAVE_TDM
  unsigned long long xbase = (unsigned long long)(xtb + (size_t)rowBase * T_SEQ * 16);
  unsigned int sx_lds = (unsigned int)(size_t)&sX[0];
#endif
  size_t woff = 0;  // laundered zero: defeats LICM of weight fragments

  for (int t = 0; t < T_SEQ; ++t) {
    asm volatile("" : "+s"(woff));
    const bf16* pWih0 = Wih0p + woff;
    const bf16* pWhh0 = Whh0p + woff;
    const bf16* pWih1 = Wih1p + woff;
    const bf16* pWhh1 = Whh1p + woff;

    __syncthreads();  // all waves done reading previous sX
#if HAVE_TDM
    if (wave == 0) {
      tdm_load_x_tile(xbase + (size_t)t * 32, sx_lds);  // t*16 elems * 2B
      __builtin_amdgcn_s_wait_tensorcnt(0);
    }
#else
    {
      int r = tid >> 1, c0 = (tid & 1) * 8;
      const bf16* src = xtb + ((size_t)(rowBase + r) * T_SEQ + t) * 16 + c0;
      *(bf16x8*)(sX + r * 32 + c0) = *(const bf16x8*)src;
    }
#endif
    __syncthreads();

    // ---------------- layer 1 ----------------
    bf16x16 aX = load_a_frag(Ax, 32);
    bf16x16 aH[4];
#pragma unroll
    for (int kc = 0; kc < 4; ++kc) aH[kc] = load_a_frag(A1 + kc * 32, HID);
#pragma unroll
    for (int nt = 0; nt < 8; ++nt) {
      f32x8 acc[4];
#pragma unroll
      for (int g = 0; g < 4; ++g) {
        acc[g] = (f32x8){0.f, 0.f, 0.f, 0.f, 0.f, 0.f, 0.f, 0.f};
        bf16x16 b = load_b_packed(pWih0, 32, 0, g * 8 + nt);
        acc[g] = wmma_bf(aX, b, acc[g]);
      }
#pragma unroll
      for (int kc = 0; kc < 4; ++kc)
#pragma unroll
        for (int g = 0; g < 4; ++g) {
          bf16x16 b = load_b_packed(pWhh0, 32, kc, g * 8 + nt);
          acc[g] = wmma_bf(aH[kc], b, acc[g]);
        }
      int col = nt * 16 + n;
      float bi = sB[col], bfg = sB[128 + col], bg = sB[256 + col], bo = sB[384 + col];
#pragma unroll
      for (int j = 0; j < 8; ++j) {
        float si = sigm(acc[0][j] + bi);
        float sf = sigm(acc[1][j] + bfg);
        float tg = tanh_fast(acc[2][j] + bg);
        float so = sigm(acc[3][j] + bo);
        float c = sf * c1[nt][j] + si * tg;
        c1[nt][j] = c;
        sH1[sh1Row + j * HID + col] = (bf16)(so * tanh_fast(c));
      }
    }

    // ---------------- layer 2 ----------------
    bf16x16 aH1[4], aH2[4];
#pragma unroll
    for (int kc = 0; kc < 4; ++kc) {
      aH1[kc] = load_a_frag(A1 + kc * 32, HID);   // x_t for layer 2 = h1(t)
      aH2[kc] = load_a_frag(A2s + kc * 32, HID);  // previous h2
    }
#pragma unroll
    for (int nt = 0; nt < 8; ++nt) {
      f32x8 acc[4];
#pragma unroll
      for (int g = 0; g < 4; ++g) acc[g] = (f32x8){0.f, 0.f, 0.f, 0.f, 0.f, 0.f, 0.f, 0.f};
#pragma unroll
      for (int kc = 0; kc < 4; ++kc)
#pragma unroll
        for (int g = 0; g < 4; ++g) {
          bf16x16 bx = load_b_packed(pWih1, 32, kc, g * 8 + nt);
          acc[g] = wmma_bf(aH1[kc], bx, acc[g]);
          bf16x16 bh = load_b_packed(pWhh1, 32, kc, g * 8 + nt);
          acc[g] = wmma_bf(aH2[kc], bh, acc[g]);
        }
      int col = nt * 16 + n;
      float bi = sB[512 + col], bfg = sB[640 + col], bg = sB[768 + col], bo = sB[896 + col];
#pragma unroll
      for (int j = 0; j < 8; ++j) {
        float si = sigm(acc[0][j] + bi);
        float sf = sigm(acc[1][j] + bfg);
        float tg = tanh_fast(acc[2][j] + bg);
        float so = sigm(acc[3][j] + bo);
        float c = sf * c2[nt][j] + si * tg;
        c2[nt][j] = c;
        sH2[sh2Row + j * HID + col] = (bf16)(so * tanh_fast(c));
      }
    }
  }

  // epilogue: enc_temporal = h2_last @ Wproj + bproj
  bf16x16 aP[4];
#pragma unroll
  for (int kc = 0; kc < 4; ++kc) aP[kc] = load_a_frag(A2s + kc * 32, HID);
#pragma unroll
  for (int nt = 0; nt < 4; ++nt) {
    f32x8 acc = {0.f, 0.f, 0.f, 0.f, 0.f, 0.f, 0.f, 0.f};
#pragma unroll
    for (int kc = 0; kc < 4; ++kc) {
      bf16x16 b = load_b_packed(Wprojp, 4, kc, nt);
      acc = wmma_bf(aP[kc], b, acc);
    }
    int col = nt * 16 + n;
    float bb = bproj[col];
#pragma unroll
    for (int j = 0; j < 8; ++j) {
      int rowg = rowBase + wave * 16 + j + 8 * hh;
      float v = acc[j] + bb;
      enc_out[(size_t)rowg * LAT + col] = v;
      enc_bf[(size_t)rowg * LAT + col] = (bf16)v;
    }
  }
}

// ---------------------------------------------------------------------------
// Flows: z <- tanh(z @ Wf + bf) x3, log_det += sum log(1-z^2) + log|det W|
// ---------------------------------------------------------------------------
__global__ __launch_bounds__(128) void flows_kernel(
    const bf16* __restrict__ enc_bf, const bf16* __restrict__ Wfp,  // packed [3][64x64]
    const float* __restrict__ fb0, const float* __restrict__ fb1,
    const float* __restrict__ fb2, const float* __restrict__ logdetW,  // [3]
    float* __restrict__ out_trans, bf16* __restrict__ z_bf,
    float* __restrict__ out_ld) {
  __shared__ __align__(16) bf16 sZ[64 * LAT];
  __shared__ __align__(16) float sZf[64 * LAT];
  __shared__ float sLD[64];
  int tid = threadIdx.x, wave = tid >> 5, lane = tid & 31;
  int rowBase = blockIdx.x * 64;
  int n = lane & 15, hh = lane >> 4;

  for (int i = tid; i < 64 * LAT; i += 128) sZ[i] = enc_bf[(size_t)rowBase * LAT + i];
  for (int i = tid; i < 64; i += 128) sLD[i] = 0.f;
  __syncthreads();

  const bf16* Az = sZ + (wave * 16) * LAT;
  for (int f = 0; f < 3; ++f) {
    const bf16* W = Wfp + (size_t)f * LAT * LAT;
    const float* fb = (f == 0) ? fb0 : ((f == 1) ? fb1 : fb2);
    bf16x16 a0 = load_a_frag(Az, LAT);
    bf16x16 a1 = load_a_frag(Az + 32, LAT);
#pragma unroll
    for (int nt = 0; nt < 4; ++nt) {
      f32x8 acc = {0.f, 0.f, 0.f, 0.f, 0.f, 0.f, 0.f, 0.f};
      acc = wmma_bf(a0, load_b_packed(W, 4, 0, nt), acc);
      acc = wmma_bf(a1, load_b_packed(W, 4, 1, nt), acc);
      int col = nt * 16 + n;
      float bb = fb[col];
#pragma unroll
      for (int j = 0; j < 8; ++j) {
        float z = tanh_fast(acc[j] + bb);
        sZf[(wave * 16 + j + 8 * hh) * LAT + col] = z;
      }
    }
    float ldw = logdetW[f];
    if (lane < 16) {
      int row = wave * 16 + lane;
      float s = 0.f;
      for (int c = 0; c < LAT; c += 4) {
        float4 z4 = *(const float4*)(sZf + row * LAT + c);
        s += __logf(1.f - z4.x * z4.x) + __logf(1.f - z4.y * z4.y) +
             __logf(1.f - z4.z * z4.z) + __logf(1.f - z4.w * z4.w);
        bf16x4 pk;
        pk[0] = (bf16)z4.x; pk[1] = (bf16)z4.y; pk[2] = (bf16)z4.z; pk[3] = (bf16)z4.w;
        *(bf16x4*)(sZ + row * LAT + c) = pk;
      }
      sLD[row] += s + ldw;
    }
  }
  __syncthreads();
  for (int i = tid; i < 64 * LAT; i += 128) {
    out_trans[(size_t)rowBase * LAT + i] = sZf[i];
    z_bf[(size_t)rowBase * LAT + i] = sZ[i];
  }
  for (int i = tid; i < 64; i += 128) out_ld[rowBase + i] = sLD[i];
}

// ---------------------------------------------------------------------------
// Host orchestration
// ---------------------------------------------------------------------------
extern "C" void kernel_launch(void* const* d_in, const int* in_sizes, int n_in,
                              void* d_out, int out_size, void* d_ws, size_t ws_size,
                              hipStream_t stream) {
  (void)in_sizes; (void)n_in; (void)out_size; (void)ws_size;

  // ---- input leaf order (jax tree_flatten: alphabetical dict keys) ----
  const float* xIn[3] = {(const float*)d_in[1], (const float*)d_in[0],
                         (const float*)d_in[2]};  // orbital, physical, signature
  const float* xt = (const float*)d_in[3];
  const int encIdx[3] = {20, 24, 28};  // orbital, physical, signature
  const int decIdx[4] = {4, 8, 12, 16};
  const int scIdx[4]  = {46, 50, 54, 58};
  const int D[4]    = {6, 32, 256, 16};   // orbital, physical, signature, temporal
  const int Dpad[4] = {16, 32, 256, 16};
  const int Kenc[3] = {32, 32, 256};

  float* out = (float*)d_out;
  const size_t SLOT = (size_t)B_TOT * LAT;
  const size_t OUT_SCORE = 0;
  const size_t OUT_ENC   = 4 * (size_t)B_TOT;
  const size_t OUT_LD    = OUT_ENC + 4 * SLOT;
  const size_t OUT_RECON = OUT_LD + 4 * (size_t)B_TOT;
  size_t reconOff[4];
  reconOff[0] = OUT_RECON;
  for (int i = 1; i < 4; ++i) reconOff[i] = reconOff[i - 1] + (size_t)B_TOT * D[i - 1];
  const size_t OUT_TRANS = reconOff[3] + (size_t)B_TOT * D[3];

  // ---- workspace bump allocator (~60 MB) ----
  char* wsb = (char*)d_ws;
  size_t off = 0;
  auto takeB = [&](size_t elems) -> bf16* {
    off = (off + 255) & ~(size_t)255; bf16* p = (bf16*)(wsb + off); off += elems * sizeof(bf16); return p;
  };
  auto takeF = [&](size_t elems) -> float* {
    off = (off + 255) & ~(size_t)255; float* p = (float*)(wsb + off); off += elems * sizeof(float); return p;
  };

  bf16* wEnc1[3]; bf16* wEnc2[3];
  for (int k = 0; k < 3; ++k) { wEnc1[k] = takeB((size_t)Kenc[k] * HID); wEnc2[k] = takeB((size_t)HID * LAT); }
  bf16* wIh0  = takeB(32 * 512);
  bf16* wHh0  = takeB(HID * 512);
  bf16* wIh1  = takeB(HID * 512);
  bf16* wHh1  = takeB(HID * 512);
  bf16* wProj = takeB(HID * LAT);
  bf16* wFlow = takeB(3 * LAT * LAT);
  bf16* wDec1[4]; bf16* wDec2[4]; bf16* wSc1[4]; bf16* wSc2[4];
  for (int k = 0; k < 4; ++k) {
    wDec1[k] = takeB((size_t)LAT * HID);
    wDec2[k] = takeB((size_t)HID * Dpad[k]);
    wSc1[k]  = takeB((size_t)LAT * HID);
    wSc2[k]  = takeB((size_t)HID * 16);
  }
  bf16* xbf[3];
  xbf[0] = takeB((size_t)B_TOT * 32);
  xbf[1] = takeB((size_t)B_TOT * 32);
  xbf[2] = takeB((size_t)B_TOT * 256);
  bf16* xtb = takeB((size_t)B_TOT * T_SEQ * 16);  // bf16 copy of x_temporal
  bf16* encbf[4]; bf16* zbf[4];
  for (int k = 0; k < 4; ++k) { encbf[k] = takeB(SLOT); zbf[k] = takeB(SLOT); }
  float* bias0 = takeF(512);
  float* bias1 = takeF(512);
  float* ldw   = takeF(3);

  auto convA = [&](const void* src, bf16* dst, int rows, int cols, int dR, int dC) {
    int total = dR * dC;
    convert_pad_kernel<<<(total + 255) / 256, 256, 0, stream>>>((const float*)src, dst,
                                                                rows, cols, dR, dC);
  };
  auto packB = [&](const void* src, bf16* dst, int rows, int cols, int Kpad, int Npad) {
    int total = (Kpad / 32) * (Npad / 16) * 32;
    pack_b_kernel<<<(total + 255) / 256, 256, 0, stream>>>((const float*)src, dst,
                                                           rows, cols, Kpad, Npad);
  };

  // ---- weight packing (B-side, fragment-ready) ----
  packB(d_in[encIdx[0]], wEnc1[0], 6,   HID, 32,  HID);
  packB(d_in[encIdx[1]], wEnc1[1], 32,  HID, 32,  HID);
  packB(d_in[encIdx[2]], wEnc1[2], 256, HID, 256, HID);
  for (int k = 0; k < 3; ++k) packB(d_in[encIdx[k] + 2], wEnc2[k], HID, LAT, HID, LAT);
  packB(d_in[39], wIh0, 16,  512, 32,  512);
  packB(d_in[38], wHh0, HID, 512, HID, 512);
  packB(d_in[43], wIh1, HID, 512, HID, 512);
  packB(d_in[42], wHh1, HID, 512, HID, 512);
  packB(d_in[62], wProj, HID, LAT, HID, LAT);
  for (int f = 0; f < 3; ++f) packB(d_in[32 + 2 * f], wFlow + (size_t)f * LAT * LAT, LAT, LAT, LAT, LAT);
  for (int k = 0; k < 4; ++k) {
    packB(d_in[decIdx[k]],     wDec1[k], LAT, HID, LAT, HID);
    packB(d_in[decIdx[k] + 2], wDec2[k], HID, D[k], HID, Dpad[k]);
    packB(d_in[scIdx[k]],      wSc1[k],  LAT, HID, LAT, HID);
    packB(d_in[scIdx[k] + 2],  wSc2[k],  HID, 1,   HID, 16);
  }
  // ---- A-side activation conversion ----
  convA(xIn[0], xbf[0], B_TOT, 6,   B_TOT, 32);
  convA(xIn[1], xbf[1], B_TOT, 32,  B_TOT, 32);
  convA(xIn[2], xbf[2], B_TOT, 256, B_TOT, 256);
  convA(xt, xtb, B_TOT * T_SEQ, 16, B_TOT * T_SEQ, 16);  // straight bf16 copy

  add_vec_kernel<<<2, 256, 0, stream>>>((const float*)d_in[41], (const float*)d_in[40], bias0, 512);
  add_vec_kernel<<<2, 256, 0, stream>>>((const float*)d_in[45], (const float*)d_in[44], bias1, 512);
  slogdet64_kernel<<<3, 64, 0, stream>>>((const float*)d_in[32], (const float*)d_in[34],
                                         (const float*)d_in[36], ldw);

  const int NB = B_TOT / 64;  // 256 blocks

  // ---- encoders (orbital, physical, signature) ----
  mlp_kernel<1><<<NB, 128, 0, stream>>>(xbf[0], wEnc1[0], (const float*)d_in[21], wEnc2[0],
                                        (const float*)d_in[23], out + OUT_ENC + 0 * SLOT,
                                        LAT, LAT, encbf[0], 0);
  mlp_kernel<1><<<NB, 128, 0, stream>>>(xbf[1], wEnc1[1], (const float*)d_in[25], wEnc2[1],
                                        (const float*)d_in[27], out + OUT_ENC + 1 * SLOT,
                                        LAT, LAT, encbf[1], 0);
  mlp_kernel<8><<<NB, 128, 0, stream>>>(xbf[2], wEnc1[2], (const float*)d_in[29], wEnc2[2],
                                        (const float*)d_in[31], out + OUT_ENC + 2 * SLOT,
                                        LAT, LAT, encbf[2], 0);
  // ---- fused LSTM + temporal projection (encodings.temporal) ----
  lstm_fused_kernel<<<NB, 128, 0, stream>>>(xtb, wIh0, wHh0, bias0, wIh1, wHh1, bias1,
                                            wProj, (const float*)d_in[63],
                                            out + OUT_ENC + 3 * SLOT, encbf[3]);
  // ---- flows + log_det per dtype ----
  for (int k = 0; k < 4; ++k) {
    flows_kernel<<<NB, 128, 0, stream>>>(encbf[k], wFlow, (const float*)d_in[33],
                                         (const float*)d_in[35], (const float*)d_in[37],
                                         ldw, out + OUT_TRANS + (size_t)k * SLOT, zbf[k],
                                         out + OUT_LD + (size_t)k * B_TOT);
  }
  // ---- decoders + anomaly scorers ----
  for (int k = 0; k < 4; ++k) {
    mlp_kernel<2><<<NB, 128, 0, stream>>>(zbf[k], wDec1[k], (const float*)d_in[decIdx[k] + 1],
                                          wDec2[k], (const float*)d_in[decIdx[k] + 3],
                                          out + reconOff[k], D[k], Dpad[k], nullptr, 0);
    mlp_kernel<2><<<NB, 128, 0, stream>>>(zbf[k], wSc1[k], (const float*)d_in[scIdx[k] + 1],
                                          wSc2[k], (const float*)d_in[scIdx[k] + 3],
                                          out + OUT_SCORE + (size_t)k * B_TOT, 1, 16, nullptr, 1);
  }
}
